// WindowMultiHeadSelfAttention_87411174408584
// MI455X (gfx1250) — matrix operand order repaired and checked by reading, not compiled, not run on hardware
//
#include <hip/hip_runtime.h>
#include <hip/hip_bf16.h>

// ---------------------------------------------------------------------------
// Types for CDNA5 WMMA (wave32)
// ---------------------------------------------------------------------------
typedef __attribute__((ext_vector_type(16))) __bf16        v16bf;
typedef __attribute__((ext_vector_type(2)))  __bf16        v2bf;
typedef __attribute__((ext_vector_type(8)))  float         v8f;
typedef __attribute__((ext_vector_type(4)))  float         v4f;
typedef __attribute__((ext_vector_type(4)))  unsigned int  v4u;

union FragCvt { v4u q[2]; v16bf v; };

#define WINDOW   7
#define P_TOK    49
#define PPAD     64
#define NUM_H    16
#define HD       32
#define DIM      512
#define B_WIN    2048
#define NW       64
#define QKV_N    1536
#define M_ROWS   (B_WIN * P_TOK)   // 100352

// ---------------------------------------------------------------------------
// bf16 conversion helpers
// ---------------------------------------------------------------------------
__device__ __forceinline__ unsigned short f2bf(float f) {
    union { float f; unsigned int u; } x; x.f = f;
    unsigned int u = x.u;
    u += 0x7FFFu + ((u >> 16) & 1u);
    return (unsigned short)(u >> 16);
}

#if __has_builtin(__builtin_amdgcn_cvt_pk_bf16_f32)
__device__ __forceinline__ unsigned int pk_bf(float a, float b) {
    union { v2bf v; unsigned int u; } x;
    x.v = __builtin_amdgcn_cvt_pk_bf16_f32(a, b);
    return x.u;
}
#else
__device__ __forceinline__ unsigned int pk_bf(float a, float b) {
    return (unsigned int)f2bf(a) | ((unsigned int)f2bf(b) << 16);
}
#endif

__device__ __forceinline__ v8f wmma_bf16(v16bf a, v16bf b, v8f c) {
    return __builtin_amdgcn_wmma_f32_16x16x32_bf16(
        false, a, false, b, (short)0, c, false, false);
}

// A-matrix 16x32 bf16 fragment (ISA 7.12.2)
__device__ __forceinline__ v16bf ldsFragA(const unsigned short* base, int strideH,
                                          int mBase, int kBase, int lane) {
    int m = mBase + (lane & 15);
    int g = lane >> 4;
    const unsigned short* p0 = base + m * strideH + kBase + 8 * g;
    FragCvt u;
    u.q[0] = *(const v4u*)p0;
    u.q[1] = *(const v4u*)(p0 + 16);
    return u.v;
}

// B-matrix 32x16 fragment from [n][k] LDS
__device__ __forceinline__ v16bf ldsFragB(const unsigned short* base, int strideH,
                                          int nBase, int kBase, int lane) {
    int n = nBase + (lane & 15);
    int g = lane >> 4;
    const unsigned short* p0 = base + n * strideH + kBase + 8 * g;
    FragCvt u;
    u.q[0] = *(const v4u*)p0;
    u.q[1] = *(const v4u*)(p0 + 16);
    return u.v;
}

// Issue async global->LDS copies of a 128x32-half tile (rows at stride DIM).
// 512 x 16B chunks across 256 threads -> 2 async instructions per thread.
__device__ __forceinline__ void asyncTile128x32(const unsigned short* gbase,
                                                int row0, int k0,
                                                unsigned short* lds, int tid) {
    for (int c = tid; c < 512; c += 256) {
        int row = c >> 2, c8 = (c & 3) * 8;       // 8 halves = 16 bytes
        unsigned int goff = (unsigned int)(((row0 + row) * DIM + k0 + c8) * 2);
        unsigned int loff = (unsigned int)(uintptr_t)(lds + row * 32 + c8);
        asm volatile("global_load_async_to_lds_b128 %0, %1, %2"
                     :: "v"(loff), "v"(goff), "s"(gbase) : "memory");
    }
}

// ---------------------------------------------------------------------------
// Kernel 0a: x (f32) -> Xbf (bf16), 8 elems/thread, b128 in / b128 out
// ---------------------------------------------------------------------------
__global__ __launch_bounds__(256)
void cvt_x_kernel(const float* __restrict__ in, unsigned short* __restrict__ out) {
    int t = blockIdx.x * 256 + threadIdx.x;
    int base = t * 8;
    v4f f0 = *(const v4f*)(in + base);
    v4f f1 = *(const v4f*)(in + base + 4);
    v4u p;
    p.x = pk_bf(f0.x, f0.y); p.y = pk_bf(f0.z, f0.w);
    p.z = pk_bf(f1.x, f1.y); p.w = pk_bf(f1.z, f1.w);
    *(v4u*)(out + base) = p;
}

// ---------------------------------------------------------------------------
// Kernel 0b: W [512][N] f32 -> Wt [N][512] bf16 (coalesced reads)
// ---------------------------------------------------------------------------
__global__ __launch_bounds__(256)
void cvt_wt_kernel(const float* __restrict__ in, unsigned short* __restrict__ out,
                   int N) {
    int t = blockIdx.x * 256 + threadIdx.x;
    int k = t / N, n = t - k * N;
    out[n * DIM + k] = f2bf(in[t]);
}

// ---------------------------------------------------------------------------
// Kernel 1: qkv = Xbf @ Wqkv_t^T + b_qkv -> scatter Q (pre-scaled), K, V
// 256 thr = 8 waves, block tile 128x128, wave tile 32x64.
// Double-buffered async LDS staging for both A and B.
// ---------------------------------------------------------------------------
__global__ __launch_bounds__(256)
void qkv_gemm_kernel(const unsigned short* __restrict__ Xbf,
                     const unsigned short* __restrict__ Wt,
                     const float* __restrict__ bias,
                     unsigned short* __restrict__ Qb,
                     unsigned short* __restrict__ Kb,
                     unsigned short* __restrict__ Vb) {
    __shared__ unsigned short sA[2][128 * 32];
    __shared__ unsigned short sB[2][128 * 32];   // [n][k]

    const int m0 = blockIdx.x * 128;
    const int n0 = blockIdx.y * 128;
    const int tid = threadIdx.x;
    const int wave = tid >> 5;
    const int lane = tid & 31;
    const int wm = (wave & 3) * 32;
    const int wn = (wave >> 2) * 64;

    v8f acc[2][4];
    for (int i = 0; i < 2; ++i)
        for (int j = 0; j < 4; ++j)
            acc[i][j] = (v8f){0.f,0.f,0.f,0.f,0.f,0.f,0.f,0.f};

    asyncTile128x32(Xbf, m0, 0, sA[0], tid);
    asyncTile128x32(Wt,  n0, 0, sB[0], tid);
    asm volatile("s_wait_asynccnt 0" ::: "memory");
    __syncthreads();

    const int NITER = DIM / 32;
    for (int it = 0; it < NITER; ++it) {
        const int cur = it & 1;
        if (it + 1 < NITER) {
            asyncTile128x32(Xbf, m0, 32 * (it + 1), sA[1 - cur], tid);
            asyncTile128x32(Wt,  n0, 32 * (it + 1), sB[1 - cur], tid);
        }
        v16bf a0 = ldsFragA(sA[cur], 32, wm,      0, lane);
        v16bf a1 = ldsFragA(sA[cur], 32, wm + 16, 0, lane);
        for (int nt = 0; nt < 4; ++nt) {
            v16bf bf = ldsFragB(sB[cur], 32, wn + 16 * nt, 0, lane);
            acc[0][nt] = wmma_bf16(a0, bf, acc[0][nt]);
            acc[1][nt] = wmma_bf16(a1, bf, acc[1][nt]);
        }
        asm volatile("s_wait_asynccnt 0" ::: "memory");
        __syncthreads();
    }

    const float qscale = 0.17677669529663687f;   // 1/sqrt(32)
    const int g = lane >> 4;
    for (int mt = 0; mt < 2; ++mt)
        for (int nt = 0; nt < 4; ++nt)
            for (int r = 0; r < 8; ++r) {
                int m = m0 + wm + 16 * mt + r + 8 * g;
                int n = n0 + wn + 16 * nt + (lane & 15);
                float v = acc[mt][nt][r] + bias[n];
                int sect = n >> 9;
                int nn = n & 511;
                int h = nn >> 5, dd = nn & 31;
                int b = m / 49, p = m - b * 49;
                int idx = (((b * NUM_H) + h) * P_TOK + p) * HD + dd;
                if (sect == 0)      Qb[idx] = f2bf(v * qscale);
                else if (sect == 1) Kb[idx] = f2bf(v);
                else                Vb[idx] = f2bf(v);
            }
}

// ---------------------------------------------------------------------------
// Kernel 2: per (window b, head h) attention. One wave per block.
// ---------------------------------------------------------------------------
__global__ __launch_bounds__(32)
void attn_kernel(const unsigned short* __restrict__ Qb,
                 const unsigned short* __restrict__ Kb,
                 const unsigned short* __restrict__ Vb,
                 const float* __restrict__ mask,
                 const float* __restrict__ relTable,
                 unsigned short* __restrict__ Abuf) {
    __shared__ unsigned short sQ[PPAD * HD];       // [p][d]
    __shared__ unsigned short sK[PPAD * HD];       // [q][d]
    __shared__ unsigned short sVt[HD * PPAD];      // [d][q]
    __shared__ unsigned short sP[PPAD * PPAD];     // [p][q] bf16
    __shared__ float          sS[PPAD * PPAD];

    const int b = blockIdx.x;
    const int h = blockIdx.y;
    const int lane = threadIdx.x;
    const int g = lane >> 4;

    const unsigned short* Qg = Qb + ((b * NUM_H) + h) * P_TOK * HD;
    const unsigned short* Kg = Kb + ((b * NUM_H) + h) * P_TOK * HD;
    const unsigned short* Vg = Vb + ((b * NUM_H) + h) * P_TOK * HD;

    for (int i = lane; i < 196; i += 32) {         // 49*32 halves = 196 x b128
        *(v4u*)(sQ + i * 8) = *(const v4u*)(Qg + i * 8);
        *(v4u*)(sK + i * 8) = *(const v4u*)(Kg + i * 8);
    }
    for (int i = 1568 + lane; i < PPAD * HD; i += 32) { sQ[i] = 0; sK[i] = 0; }
    for (int i = lane; i < P_TOK * 16; i += 32) {  // V transpose [p][d]->[d][q]
        int p = i >> 4, d2 = (i & 15) * 2;
        unsigned int u = *(const unsigned int*)(Vg + p * HD + d2);
        sVt[(d2 + 0) * PPAD + p] = (unsigned short)u;
        sVt[(d2 + 1) * PPAD + p] = (unsigned short)(u >> 16);
    }
    for (int i = lane; i < HD * 15; i += 32) {
        int d = i / 15, p = P_TOK + i % 15;
        sVt[d * PPAD + p] = 0;
    }
    __syncthreads();

    const float* mrow = mask + (b & (NW - 1)) * P_TOK * P_TOK;

    // ---- S = Q * K^T ----
    for (int mt = 0; mt < 4; ++mt) {
        v16bf a = ldsFragA(sQ, HD, 16 * mt, 0, lane);
        for (int nt = 0; nt < 4; ++nt) {
            v16bf bf = ldsFragB(sK, HD, 16 * nt, 0, lane);
            v8f c = (v8f){0.f,0.f,0.f,0.f,0.f,0.f,0.f,0.f};
            c = wmma_bf16(a, bf, c);
            int col = 16 * nt + (lane & 15);
            for (int r = 0; r < 8; ++r) {
                int row = 16 * mt + r + 8 * g;
                float v = c[r];
                if (col < P_TOK) {
                    if (row < P_TOK) {
                        int pi = row / WINDOW, pj = row - pi * WINDOW;
                        int qi = col / WINDOW, qj = col - qi * WINDOW;
                        int rel = (pi - qi + WINDOW - 1) * (2 * WINDOW - 1)
                                + (pj - qj + WINDOW - 1);
                        v += relTable[rel * NUM_H + h] + mrow[row * P_TOK + col];
                    }
                } else {
                    v = -1e30f;
                }
                sS[row * PPAD + col] = v;
            }
        }
    }
    __syncthreads();

    // ---- softmax (each lane: 2 rows) ----
    for (int rr = 0; rr < 2; ++rr) {
        int row = lane + 32 * rr;
        const float* s = sS + row * PPAD;
        float mx = -1e30f;
        for (int c2 = 0; c2 < PPAD; ++c2) mx = fmaxf(mx, s[c2]);
        float sum = 0.f;
        for (int c2 = 0; c2 < PPAD; ++c2) sum += __expf(s[c2] - mx);
        float inv = 1.f / sum;
        unsigned short* pr = sP + row * PPAD;
        for (int c2 = 0; c2 < PPAD; ++c2) pr[c2] = f2bf(__expf(s[c2] - mx) * inv);
    }
    __syncthreads();

    // ---- O = P * V ----
    for (int mt = 0; mt < 4; ++mt)
        for (int nt = 0; nt < 2; ++nt) {
            v8f c = (v8f){0.f,0.f,0.f,0.f,0.f,0.f,0.f,0.f};
            for (int kt = 0; kt < 2; ++kt) {
                v16bf a  = ldsFragA(sP,  PPAD, 16 * mt, 32 * kt, lane);
                v16bf bf = ldsFragB(sVt, PPAD, 16 * nt, 32 * kt, lane);
                c = wmma_bf16(a, bf, c);
            }
            int col = 16 * nt + (lane & 15);
            for (int r = 0; r < 8; ++r) {
                int row = 16 * mt + r + 8 * g;
                if (row < P_TOK)
                    Abuf[(b * P_TOK + row) * DIM + h * HD + col] = f2bf(c[r]);
            }
        }
}

// ---------------------------------------------------------------------------
// Kernel 3: out = Abuf(bf16) @ Wproj_t^T + b_proj (f32 out)
// ---------------------------------------------------------------------------
__global__ __launch_bounds__(256)
void proj_gemm_kernel(const unsigned short* __restrict__ A,
                      const unsigned short* __restrict__ Wt,
                      const float* __restrict__ bias,
                      float* __restrict__ out) {
    __shared__ unsigned short sA[2][128 * 32];
    __shared__ unsigned short sB[2][128 * 32];   // [n][k]

    const int m0 = blockIdx.x * 128;
    const int n0 = blockIdx.y * 128;
    const int tid = threadIdx.x;
    const int wave = tid >> 5;
    const int lane = tid & 31;
    const int wm = (wave & 3) * 32;
    const int wn = (wave >> 2) * 64;

    v8f acc[2][4];
    for (int i = 0; i < 2; ++i)
        for (int j = 0; j < 4; ++j)
            acc[i][j] = (v8f){0.f,0.f,0.f,0.f,0.f,0.f,0.f,0.f};

    asyncTile128x32(A,  m0, 0, sA[0], tid);
    asyncTile128x32(Wt, n0, 0, sB[0], tid);
    asm volatile("s_wait_asynccnt 0" ::: "memory");
    __syncthreads();

    const int NITER = DIM / 32;
    for (int it = 0; it < NITER; ++it) {
        const int cur = it & 1;
        if (it + 1 < NITER) {
            asyncTile128x32(A,  m0, 32 * (it + 1), sA[1 - cur], tid);
            asyncTile128x32(Wt, n0, 32 * (it + 1), sB[1 - cur], tid);
        }
        v16bf a0 = ldsFragA(sA[cur], 32, wm,      0, lane);
        v16bf a1 = ldsFragA(sA[cur], 32, wm + 16, 0, lane);
        for (int nt = 0; nt < 4; ++nt) {
            v16bf bf = ldsFragB(sB[cur], 32, wn + 16 * nt, 0, lane);
            acc[0][nt] = wmma_bf16(a0, bf, acc[0][nt]);
            acc[1][nt] = wmma_bf16(a1, bf, acc[1][nt]);
        }
        asm volatile("s_wait_asynccnt 0" ::: "memory");
        __syncthreads();
    }

    const int g = lane >> 4;
    for (int mt = 0; mt < 2; ++mt)
        for (int nt = 0; nt < 4; ++nt)
            for (int r = 0; r < 8; ++r) {
                int m = m0 + wm + 16 * mt + r + 8 * g;
                int n = n0 + wn + 16 * nt + (lane & 15);
                out[m * DIM + n] = acc[mt][nt][r] + bias[n];
            }
}

// ---------------------------------------------------------------------------
extern "C" void kernel_launch(void* const* d_in, const int* in_sizes, int n_in,
                              void* d_out, int out_size, void* d_ws, size_t ws_size,
                              hipStream_t stream) {
    const float* x        = (const float*)d_in[0];
    const float* mask     = (const float*)d_in[1];
    const float* w_qkv    = (const float*)d_in[2];
    const float* b_qkv    = (const float*)d_in[3];
    const float* w_proj   = (const float*)d_in[4];
    const float* b_proj   = (const float*)d_in[5];
    const float* relTable = (const float*)d_in[6];

    const size_t headElems = (size_t)B_WIN * NUM_H * P_TOK * HD; // 51,380,224
    unsigned short* Qb      = (unsigned short*)d_ws;
    unsigned short* Kb      = Qb + headElems;
    unsigned short* Vb      = Kb + headElems;
    unsigned short* Abuf    = Vb + headElems;
    unsigned short* Xbf     = Abuf + headElems;
    unsigned short* Wqkv_t  = Xbf + headElems;                 // [1536][512]
    unsigned short* Wproj_t = Wqkv_t + (size_t)QKV_N * DIM;    // [512][512]

    // 0) one-time conversions / weight transposes (bf16)
    cvt_x_kernel<<<(M_ROWS * DIM) / (256 * 8), 256, 0, stream>>>(x, Xbf);
    cvt_wt_kernel<<<(DIM * QKV_N) / 256, 256, 0, stream>>>(w_qkv, Wqkv_t, QKV_N);
    cvt_wt_kernel<<<(DIM * DIM) / 256, 256, 0, stream>>>(w_proj, Wproj_t, DIM);

    // 1) QKV GEMM
    qkv_gemm_kernel<<<dim3(M_ROWS / 128, QKV_N / 128), 256, 0, stream>>>(
        Xbf, Wqkv_t, b_qkv, Qb, Kb, Vb);

    // 2) attention
    attn_kernel<<<dim3(B_WIN, NUM_H), 32, 0, stream>>>(
        Qb, Kb, Vb, mask, relTable, Abuf);

    // 3) proj GEMM -> f32 out
    proj_gemm_kernel<<<dim3(M_ROWS / 128, DIM / 128), 256, 0, stream>>>(
        Abuf, Wproj_t, b_proj, (float*)d_out);
}
